// HighClue_Guided_Intra_and_Cross_Domain_Refin_47579647705582
// MI455X (gfx1250) — matrix operand order repaired and checked by reading, not compiled, not run on hardware
//
#include <hip/hip_runtime.h>
#include <hip/hip_bf16.h>

#define NN   4
#define CC   64
#define HWN  4096
#define CNTF 16384.0f
#define EPSV 1e-5f
#define LOG2E 1.44269504088896f

typedef __attribute__((ext_vector_type(16))) _Float16 v16h;
typedef __attribute__((ext_vector_type(8)))  _Float16 v8h;
typedef __attribute__((ext_vector_type(8)))  float    v8f;

union Frag16 { v16h v; _Float16 h[16]; v8h p[2]; };

__device__ __forceinline__ float half_red_max(float v) {
#pragma unroll
  for (int d = 1; d < 16; d <<= 1) v = fmaxf(v, __shfl_xor(v, d, 32));
  return v;
}
__device__ __forceinline__ float wave_red_sum(float v) {
#pragma unroll
  for (int d = 1; d < 32; d <<= 1) v += __shfl_xor(v, d, 32);
  return v;
}

// load 8 K B-fragments (one 64-key slab), rows at krow0 (+t*16*CC), contiguous 32B/lane
__device__ __forceinline__ void load_kslab(Frag16* kf, const _Float16* krow0, int cb) {
#pragma unroll
  for (int t = 0; t < 4; ++t) {
    const _Float16* krow = krow0 + t * 16 * CC;
    kf[2*t+0].p[0] = *(const v8h*)(krow + cb);
    kf[2*t+0].p[1] = *(const v8h*)(krow + cb + 8);
    kf[2*t+1].p[0] = *(const v8h*)(krow + 32 + cb);
    kf[2*t+1].p[1] = *(const v8h*)(krow + 32 + cb + 8);
  }
}

// ---------------- kernel 0: zero BN-stat accumulators ----------------
__global__ void zero_stats_kernel(float* stats) { stats[threadIdx.x] = 0.0f; }

// ---------------- kernel 1: projections + BN stats + gated V ----------------
__global__ __launch_bounds__(256) void proj_stats_kernel(
    const float* __restrict__ x,  const float* __restrict__ h0, const float* __restrict__ h1,
    const float* __restrict__ Wq, const float* __restrict__ bq,
    const float* __restrict__ Wk, const float* __restrict__ bk,
    const float* __restrict__ Wv, const float* __restrict__ bv,
    float* __restrict__ stats, _Float16* __restrict__ Vh)
{
  __shared__ float sX[CC * 64];
  __shared__ float sWq[CC * CC];
  __shared__ float sWk[CC * CC];
  __shared__ float sWv[CC * CC];
  const int tid = threadIdx.x;
  const int n   = blockIdx.x >> 6;
  const int p0  = (blockIdx.x & 63) * 64;

  for (int i = tid; i < CC * 64; i += 256) {
    int c = i >> 6, pp = i & 63;
    sX[i] = x[((size_t)n * CC + c) * HWN + p0 + pp];
  }
  for (int i = tid; i < CC * CC; i += 256) { sWq[i] = Wq[i]; sWk[i] = Wk[i]; sWv[i] = Wv[i]; }
  __syncthreads();

  const int pp = tid & 63;
  const int og = tid >> 6;   // uniform per wave
  float aq[16], ak[16], av[16];
#pragma unroll
  for (int i = 0; i < 16; ++i) { aq[i] = bq[og*16+i]; ak[i] = bk[og*16+i]; av[i] = bv[og*16+i]; }

  for (int c = 0; c < CC; ++c) {
    float xv = sX[c * 64 + pp];
#pragma unroll
    for (int oi = 0; oi < 16; ++oi) {
      int o = og * 16 + oi;
      aq[oi] = fmaf(sWq[o * CC + c], xv, aq[oi]);
      ak[oi] = fmaf(sWk[o * CC + c], xv, ak[oi]);
      av[oi] = fmaf(sWv[o * CC + c], xv, av[oi]);
    }
  }

  const int lane = tid & 31;
#pragma unroll
  for (int oi = 0; oi < 16; ++oi) {
    int o = og * 16 + oi;
    size_t gidx = ((size_t)n * CC + o) * HWN + p0 + pp;
    float gate = h0[gidx] + h1[gidx];
    Vh[gidx] = (_Float16)(av[oi] * gate);      // gated V, f16, [n][c][p] layout

    float sq  = wave_red_sum(aq[oi]);
    float sq2 = wave_red_sum(aq[oi] * aq[oi]);
    float sk  = wave_red_sum(ak[oi]);
    float sk2 = wave_red_sum(ak[oi] * ak[oi]);
    if (lane == 0) {
      atomicAdd(&stats[o],       sq);
      atomicAdd(&stats[64 + o],  sq2);
      atomicAdd(&stats[128 + o], sk);
      atomicAdd(&stats[192 + o], sk2);
    }
  }
}

// ---------------- kernel 2: recompute Q/K proj, BN+ReLU, store f16 [p][c] ----------------
// Q additionally scaled by log2(e) so the attention kernel can use bare v_exp_f32.
__global__ __launch_bounds__(256) void bn_finalize_kernel(
    const float* __restrict__ x,
    const float* __restrict__ Wq, const float* __restrict__ bq,
    const float* __restrict__ gq, const float* __restrict__ betaq,
    const float* __restrict__ Wk, const float* __restrict__ bk,
    const float* __restrict__ gk, const float* __restrict__ betak,
    const float* __restrict__ stats,
    _Float16* __restrict__ Qh, _Float16* __restrict__ Kh)
{
  __shared__ float sX[CC * 64];
  __shared__ float sWq[CC * CC];
  __shared__ float sWk[CC * CC];
  const int tid = threadIdx.x;
  const int n   = blockIdx.x >> 6;
  const int p0  = (blockIdx.x & 63) * 64;

  for (int i = tid; i < CC * 64; i += 256) {
    int c = i >> 6, pp = i & 63;
    sX[i] = x[((size_t)n * CC + c) * HWN + p0 + pp];
  }
  for (int i = tid; i < CC * CC; i += 256) { sWq[i] = Wq[i]; sWk[i] = Wk[i]; }
  __syncthreads();

  const int pp = tid & 63;
  const int og = tid >> 6;
  float aq[16], ak[16];
#pragma unroll
  for (int i = 0; i < 16; ++i) { aq[i] = bq[og*16+i]; ak[i] = bk[og*16+i]; }

  for (int c = 0; c < CC; ++c) {
    float xv = sX[c * 64 + pp];
#pragma unroll
    for (int oi = 0; oi < 16; ++oi) {
      int o = og * 16 + oi;
      aq[oi] = fmaf(sWq[o * CC + c], xv, aq[oi]);
      ak[oi] = fmaf(sWk[o * CC + c], xv, ak[oi]);
    }
  }

  Frag16 oq, ok;
#pragma unroll
  for (int oi = 0; oi < 16; ++oi) {
    int o = og * 16 + oi;
    float muq = stats[o]       * (1.0f / CNTF);
    float vq  = stats[64 + o]  * (1.0f / CNTF) - muq * muq;
    float yq  = (aq[oi] - muq) * rsqrtf(vq + EPSV) * gq[o] + betaq[o];
    oq.h[oi]  = (_Float16)(fmaxf(yq, 0.0f) * LOG2E);   // exp2-domain Q
    float muk = stats[128 + o] * (1.0f / CNTF);
    float vk  = stats[192 + o] * (1.0f / CNTF) - muk * muk;
    float yk  = (ak[oi] - muk) * rsqrtf(vk + EPSV) * gk[o] + betak[o];
    ok.h[oi]  = (_Float16)fmaxf(yk, 0.0f);
  }
  size_t base = ((size_t)n * HWN + p0 + pp) * CC + og * 16;   // [n][p][c]
  *(v8h*)(Qh + base)     = oq.p[0];
  *(v8h*)(Qh + base + 8) = oq.p[1];
  *(v8h*)(Kh + base)     = ok.p[0];
  *(v8h*)(Kh + base + 8) = ok.p[1];
}

// ---------------- kernel 3: two-pass flash attention with WMMA ----------------
// grid 256 (n * 64 q-blocks of 64 rows), block 128 (4 waves x 16 q-rows).
__global__ __launch_bounds__(128) void attn_kernel(
    const _Float16* __restrict__ Qh, const _Float16* __restrict__ Kh,
    const _Float16* __restrict__ Vh, float* __restrict__ out)
{
  __shared__ __attribute__((aligned(32))) _Float16 sP[4][16 * 64];  // per-wave P staging
  const int tid  = threadIdx.x;
  const int wave = tid >> 5;
  const int lane = tid & 31;
  const int hsel = lane >> 4;       // lane half: 0 or 1
  const int l16  = lane & 15;
  const int n    = blockIdx.x >> 6;
  const int q0   = (blockIdx.x & 63) * 64 + wave * 16;
  const int ab   = hsel * 8;        // unused; A-frags also use cb
  const int cb   = hsel * 16;       // B-frag K-chunk base

  // Q A-fragments (c = 0..31 and 32..63), row = q0 + l16
  const _Float16* qrow = Qh + ((size_t)n * HWN + q0 + l16) * CC;
  Frag16 qa0, qa1;
  qa0.p[0] = *(const v8h*)(qrow + cb);
  qa0.p[1] = *(const v8h*)(qrow + cb + 8);
  qa1.p[0] = *(const v8h*)(qrow + 32 + cb);
  qa1.p[1] = *(const v8h*)(qrow + 32 + cb + 8);

  const _Float16* kbase = Kh + (size_t)n * HWN * CC;   // [p][c]
  const _Float16* vbase = Vh + (size_t)n * CC * HWN;   // [c][p]
  v8f zf = {};

  // ---------------- pass 1: exact per-row max (double-buffered K slabs) ----------------
  v8f vm;
#pragma unroll
  for (int r = 0; r < 8; ++r) vm[r] = -1e30f;

  {
    const _Float16* kp0 = kbase + (size_t)l16 * CC;
    const _Float16* kp  = kp0;
    Frag16 kfA[8], kfB[8];
    load_kslab(kfA, kp, cb);                       // prime slab 0
    for (int kb = 0; kb < HWN; kb += 128) {
      load_kslab(kfB, kp + 64 * CC, cb);           // odd slab in flight
      {
        v8f s[4];
#pragma unroll
        for (int t = 0; t < 4; ++t)
          s[t] = __builtin_amdgcn_wmma_f32_16x16x32_f16(false, qa0.v, false, kfA[2*t+0].v, (short)0, zf, false, false);
#pragma unroll
        for (int t = 0; t < 4; ++t)
          s[t] = __builtin_amdgcn_wmma_f32_16x16x32_f16(false, qa1.v, false, kfA[2*t+1].v, (short)0, s[t], false, false);
#pragma unroll
        for (int t = 0; t < 4; ++t)
#pragma unroll
          for (int r = 0; r < 8; ++r) vm[r] = fmaxf(vm[r], s[t][r]);
      }
      const _Float16* kpn = (kb + 128 < HWN) ? (kp + 128 * CC) : kp0;  // clamp (harmless reload)
      load_kslab(kfA, kpn, cb);                    // next even slab in flight
      __builtin_prefetch(kpn + 64 * CC, 0, 1);
      {
        v8f s[4];
#pragma unroll
        for (int t = 0; t < 4; ++t)
          s[t] = __builtin_amdgcn_wmma_f32_16x16x32_f16(false, qa0.v, false, kfB[2*t+0].v, (short)0, zf, false, false);
#pragma unroll
        for (int t = 0; t < 4; ++t)
          s[t] = __builtin_amdgcn_wmma_f32_16x16x32_f16(false, qa1.v, false, kfB[2*t+1].v, (short)0, s[t], false, false);
#pragma unroll
        for (int t = 0; t < 4; ++t)
#pragma unroll
          for (int r = 0; r < 8; ++r) vm[r] = fmaxf(vm[r], s[t][r]);
      }
      kp = kpn;
    }
  }
  v8f mneg;
#pragma unroll
  for (int r = 0; r < 8; ++r) mneg[r] = -half_red_max(vm[r]);

  // ---------------- pass 2: P = exp2(S - M); sums via P @ ones; O += P @ V ----------------
  Frag16 onesB;
#pragma unroll
  for (int i = 0; i < 16; ++i) onesB.h[i] = (_Float16)1.0f;

  v8f acc[4];
#pragma unroll
  for (int t = 0; t < 4; ++t) acc[t] = zf;
  v8f lacc = zf;
  _Float16* myP = &sP[wave][0];

  {
    const _Float16* kp0 = kbase + (size_t)l16 * CC;
    const _Float16* vp0 = vbase + (size_t)l16 * HWN + hsel * 16;
    const _Float16* kp  = kp0;
    const _Float16* vp  = vp0;
    Frag16 kfA[8], kfB[8];
    load_kslab(kfA, kp, cb);                       // prime slab 0

    for (int kb = 0; kb < HWN; kb += 128) {
      load_kslab(kfB, kp + 64 * CC, cb);           // odd slab in flight

#pragma unroll
      for (int half = 0; half < 2; ++half) {
        Frag16* kf = half ? kfB : kfA;
        const _Float16* vph = vp + half * 64;

        // -- S WMMAs (C preloaded with -rowmax) --
        v8f s[4];
#pragma unroll
        for (int t = 0; t < 4; ++t)
          s[t] = __builtin_amdgcn_wmma_f32_16x16x32_f16(false, qa0.v, false, kf[2*t+0].v, (short)0, mneg, false, false);
#pragma unroll
        for (int t = 0; t < 4; ++t)
          s[t] = __builtin_amdgcn_wmma_f32_16x16x32_f16(false, qa1.v, false, kf[2*t+1].v, (short)0, s[t], false, false);

        // refill the just-consumed K buffer with the slab two steps ahead
        if (half == 0) {
          const _Float16* kpn = (kb + 128 < HWN) ? (kp + 128 * CC) : kp0;
          load_kslab(kfA, kpn, cb);
        }

        // -- V fragments issued early: hide under exp2 + LDS transpose --
        Frag16 vf[8];
#pragma unroll
        for (int ct = 0; ct < 4; ++ct) {
          const _Float16* vrow = vph + (size_t)ct * 16 * HWN;
          vf[2*ct+0].p[0] = *(const v8h*)(vrow);
          vf[2*ct+0].p[1] = *(const v8h*)(vrow + 8);
          vf[2*ct+1].p[0] = *(const v8h*)(vrow + 32);
          vf[2*ct+1].p[1] = *(const v8h*)(vrow + 40);
        }

        // -- P = exp2(S - M), staged to LDS in C-layout --
#pragma unroll
        for (int t = 0; t < 4; ++t)
#pragma unroll
          for (int r = 0; r < 8; ++r) {
            int prow = r + hsel * 8;
            myP[prow * 64 + t * 16 + l16] = (_Float16)__builtin_amdgcn_exp2f(s[t][r]);
          }
        asm volatile("s_wait_dscnt 0x0" ::: "memory");   // in-order LDS per wave

        // -- reload P as two 16x32 A-fragments --
        Frag16 pa0, pa1;
        const _Float16* prow_ = myP + l16 * 64;
        pa0.p[0] = *(const v8h*)(prow_ + cb);
        pa0.p[1] = *(const v8h*)(prow_ + cb + 8);
        pa1.p[0] = *(const v8h*)(prow_ + 32 + cb);
        pa1.p[1] = *(const v8h*)(prow_ + 32 + cb + 8);

        // -- row sums: l += P @ ones --
        lacc = __builtin_amdgcn_wmma_f32_16x16x32_f16(false, pa0.v, false, onesB.v, (short)0, lacc, false, false);
        lacc = __builtin_amdgcn_wmma_f32_16x16x32_f16(false, pa1.v, false, onesB.v, (short)0, lacc, false, false);

        // -- O += P @ Vgated --
#pragma unroll
        for (int ct = 0; ct < 4; ++ct) {
          acc[ct] = __builtin_amdgcn_wmma_f32_16x16x32_f16(false, pa0.v, false, vf[2*ct+0].v, (short)0, acc[ct], false, false);
          acc[ct] = __builtin_amdgcn_wmma_f32_16x16x32_f16(false, pa1.v, false, vf[2*ct+1].v, (short)0, acc[ct], false, false);
        }
      }

      kp = (kb + 128 < HWN) ? (kp + 128 * CC) : kp0;
      vp += 128;
    }
  }

  // ---------------- epilogue: normalize, store transposed [n][c][q] ----------------
#pragma unroll
  for (int r = 0; r < 8; ++r) {
    float inv = 1.0f / lacc[r];
    int q = q0 + r + hsel * 8;
#pragma unroll
    for (int ct = 0; ct < 4; ++ct) {
      int c = ct * 16 + l16;
      out[((size_t)n * CC + c) * HWN + q] = acc[ct][r] * inv;
    }
  }
}

// ---------------- launch ----------------
extern "C" void kernel_launch(void* const* d_in, const int* in_sizes, int n_in,
                              void* d_out, int out_size, void* d_ws, size_t ws_size,
                              hipStream_t stream) {
  const float* x     = (const float*)d_in[0];
  const float* h0    = (const float*)d_in[1];
  const float* h1    = (const float*)d_in[2];
  const float* Wq    = (const float*)d_in[3];
  const float* bq    = (const float*)d_in[4];
  const float* gq    = (const float*)d_in[5];
  const float* betaq = (const float*)d_in[6];
  const float* Wk    = (const float*)d_in[7];
  const float* bk    = (const float*)d_in[8];
  const float* gk    = (const float*)d_in[9];
  const float* betak = (const float*)d_in[10];
  const float* Wv    = (const float*)d_in[11];
  const float* bv    = (const float*)d_in[12];
  float* out = (float*)d_out;

  char* ws = (char*)d_ws;
  const size_t FSZ = (size_t)NN * HWN * CC * sizeof(_Float16);   // 2 MiB per f16 tensor
  float*    stats = (float*)ws;                  // 256 floats
  _Float16* Qh    = (_Float16*)(ws + 4096);
  _Float16* Kh    = (_Float16*)(ws + 4096 + FSZ);
  _Float16* Vh    = (_Float16*)(ws + 4096 + 2 * FSZ);

  zero_stats_kernel<<<1, 256, 0, stream>>>(stats);
  proj_stats_kernel<<<256, 256, 0, stream>>>(x, h0, h1, Wq, bq, Wk, bk, Wv, bv, stats, Vh);
  bn_finalize_kernel<<<256, 256, 0, stream>>>(x, Wq, bq, gq, betaq, Wk, bk, gk, betak, stats, Qh, Kh);
  attn_kernel<<<256, 128, 0, stream>>>(Qh, Kh, Vh, out);
}
